// SparseQuery_65386582114945
// MI455X (gfx1250) — matrix operand order repaired and checked by reading, not compiled, not run on hardware
//
#include <hip/hip_runtime.h>
#include <math.h>

#define IN_F 1024
#define HID  256
#define NE   16
#define HD   64
#define TK   4

typedef __attribute__((ext_vector_type(16))) __bf16 bf16x16;
typedef __attribute__((ext_vector_type(8)))  __bf16 bf16x8;
typedef __attribute__((ext_vector_type(2)))  __bf16 bf16x2;
typedef __attribute__((ext_vector_type(8)))  float  f32x8;

static __device__ __forceinline__ unsigned int pack2(float a, float b) {
  bf16x2 v;
  v[0] = (__bf16)a;
  v[1] = (__bf16)b;                       // backend: v_cvt_pk_bf16_f32
  return __builtin_bit_cast(unsigned int, v);
}
static __device__ __forceinline__ f32x8 wmma_bf16(const bf16x16& a, const bf16x16& b, f32x8 c) {
  return __builtin_amdgcn_wmma_f32_16x16x32_bf16(false, a, false, b, (short)0, c, false, false);
}
static __device__ __forceinline__ bf16x16 ldsA(const __bf16* base, int row, int rowStride,
                                               int kk, int hl) {
  // contiguous A layout (matches packed B): elem j -> k = kk+16*hl+j
  bf16x8 lo = *(const bf16x8*)&base[row * rowStride + kk + 16 * hl];
  bf16x8 hi = *(const bf16x8*)&base[row * rowStride + kk + 16 * hl + 8];
  bf16x16 a;
#pragma unroll
  for (int j = 0; j < 8; ++j) { a[j] = lo[j]; a[8 + j] = hi[j]; }
  return a;
}

// ---------------------------------------------------------------------------
// Kernel 1: normalized centroids, folding in exp(temperature)/sqrt(HID)
// ---------------------------------------------------------------------------
__global__ void prep_centroids_kernel(const float* __restrict__ hc,
                                      const float* __restrict__ temp,
                                      float* __restrict__ cnn) {
  int e = threadIdx.x;
  if (e < NE) {
    float s = 0.f;
    for (int j = 0; j < HID; ++j) { float v = hc[e * HID + j]; s += v * v; }
    float inv = 1.0f / fmaxf(sqrtf(s), 1e-12f);
    float scale = inv * expf(temp[0]) * (1.0f / 16.0f);   // sqrt(256) == 16
    for (int j = 0; j < HID; ++j) cnn[e * HID + j] = hc[e * HID + j] * scale;
  }
}

// ---------------------------------------------------------------------------
// Kernel 2: pack an f32 [numMat][K][N] weight into the WMMA B-fragment layout
// dst groups of 16 bf16: g = ((mat*K/32 + kk32)*N + n)*2 + hl
// lane fragment element j <-> W[mat][kk32*32 + 16*hl + j][n]
// ---------------------------------------------------------------------------
__global__ void pack_b_kernel(const float* __restrict__ src, __bf16* __restrict__ dst,
                              int K32, int N, int total) {
  int g = blockIdx.x * 256 + threadIdx.x;
  if (g >= total) return;
  int hl = g & 1;
  int t  = g >> 1;
  int n  = t % N;
  int r  = t / N;
  int kk32 = r % K32;
  int mat  = r / K32;
  const float* s = src + ((size_t)mat * K32 * 32 + kk32 * 32 + 16 * hl) * N + n;
  bf16x16 v;
#pragma unroll
  for (int j = 0; j < 16; ++j) v[j] = (__bf16)s[(size_t)j * N];
  *(bf16x16*)&dst[(size_t)g * 16] = v;
}

// ---------------------------------------------------------------------------
// Kernel 3: router GEMM (WMMA bf16) + cosine logits (WMMA) + softmax + top-4
// block = 256 threads (8 waves). M-tile = 16 tokens, each wave does 2 N-tiles.
// ---------------------------------------------------------------------------
__global__ void router_topk_kernel(const float* __restrict__ x,
                                   const __bf16* __restrict__ rwPk,
                                   const __bf16* __restrict__ cnnPk,
                                   int* __restrict__ topE,
                                   float* __restrict__ topG) {
  __shared__ __attribute__((aligned(16))) __bf16 Abf[16 * IN_F]; // 32 KB
  __shared__ float zs[16 * HID];                                 // 16 KB (f32 z, for norms)
  __shared__ __attribute__((aligned(16))) __bf16 zbf[16 * HID];  // 8 KB  (bf16 z, logit GEMM A)
  __shared__ float part[256];
  __shared__ float invn[16];
  __shared__ float logit[16 * NE];

  const int tid  = threadIdx.x;
  const int tok0 = blockIdx.x * 16;

  // ---- stage A tile (16 tokens x 1024) as bf16 in LDS ----
  {
    const int row  = tid >> 4;     // 0..15
    const int part_ = tid & 15;    // 64 floats each
    const float4* src = (const float4*)(x + (size_t)(tok0 + row) * IN_F + part_ * 64);
    unsigned int* dst = (unsigned int*)&Abf[row * IN_F + part_ * 64];
#pragma unroll
    for (int i = 0; i < 16; ++i) {
      float4 v = src[i];
      dst[i * 2 + 0] = pack2(v.x, v.y);
      dst[i * 2 + 1] = pack2(v.z, v.w);
    }
  }
  __syncthreads();

  const int w    = tid >> 5;       // wave 0..7
  const int lane = tid & 31;
  const int hl   = lane >> 4;
  const int ln   = lane & 15;

  const bf16x16* BW = (const bf16x16*)rwPk;
  f32x8 c0 = {};
  f32x8 c1 = {};
  for (int kk = 0; kk < IN_F; kk += 32) {
    const int kk32 = kk >> 5;
    bf16x16 a = ldsA(Abf, ln, IN_F, kk, hl);
    const int n0 = w * 32 + ln;
    bf16x16 b0 = BW[((size_t)kk32 * HID + n0) * 2 + hl];
    bf16x16 b1 = BW[((size_t)kk32 * HID + n0 + 16) * 2 + hl];
    if (kk + 32 < IN_F)
      __builtin_prefetch(&BW[((size_t)(kk32 + 1) * HID + n0) * 2 + hl], 0, 0);
    c0 = wmma_bf16(a, b0, c0);
    c1 = wmma_bf16(a, b1, c1);
  }

  // ---- spill z (f32 for norms, bf16 for the logit GEMM) ----
#pragma unroll
  for (int r = 0; r < 8; ++r) {
    int m = r + 8 * hl;
    zs[m * HID + w * 32 + ln]       = c0[r];
    zs[m * HID + w * 32 + 16 + ln]  = c1[r];
    zbf[m * HID + w * 32 + ln]      = (__bf16)c0[r];
    zbf[m * HID + w * 32 + 16 + ln] = (__bf16)c1[r];
  }
  __syncthreads();

  // ---- row norms: 256-way partial + 16-way reduce ----
  {
    int m = tid >> 4, ch = tid & 15;
    const float4* zp = (const float4*)&zs[m * HID + ch * 16];
    float s = 0.f;
#pragma unroll
    for (int i = 0; i < 4; ++i) {
      float4 v = zp[i];
      s += v.x * v.x + v.y * v.y + v.z * v.z + v.w * v.w;
    }
    part[tid] = s;
  }
  __syncthreads();
  if (tid < 16) {
    float s = 0.f;
#pragma unroll
    for (int i = 0; i < 16; ++i) s += part[tid * 16 + i];
    invn[tid] = 1.0f / fmaxf(sqrtf(s), 1e-12f);
  }
  __syncthreads();

  // ---- logits = (z @ cnn^T) * invn[m] via WMMA, wave 0 only ----
  if (w == 0) {
    const bf16x16* BC = (const bf16x16*)cnnPk;
    f32x8 cl = {};
#pragma unroll
    for (int kk32 = 0; kk32 < HID / 32; ++kk32) {
      bf16x16 a = ldsA(zbf, ln, HID, kk32 * 32, hl);
      bf16x16 b = BC[((size_t)kk32 * NE + ln) * 2 + hl];
      cl = wmma_bf16(a, b, cl);
    }
#pragma unroll
    for (int r = 0; r < 8; ++r) {
      int m = r + 8 * hl;
      logit[m * NE + ln] = cl[r] * invn[m];
    }
  }
  __syncthreads();

  if (tid < 16) { // softmax over 16 + top-4 (ties -> lowest index, like top_k)
    float p[NE];
    float mx = -1e30f;
    for (int e = 0; e < NE; ++e) mx = fmaxf(mx, logit[tid * NE + e]);
    float s = 0.f;
    for (int e = 0; e < NE; ++e) { float v = expf(logit[tid * NE + e] - mx); p[e] = v; s += v; }
    float inv = 1.0f / s;
    for (int e = 0; e < NE; ++e) p[e] *= inv;
    int tok = tok0 + tid;
    for (int k = 0; k < TK; ++k) {
      int bi = 0; float bv = p[0];
      for (int e = 1; e < NE; ++e) if (p[e] > bv) { bv = p[e]; bi = e; }
      topE[tok * TK + k] = bi;
      topG[tok * TK + k] = bv;
      p[bi] = -1.f;
    }
  }
}

// ---------------------------------------------------------------------------
// Kernels 4-6: stable counting sort of (token,slot) rows by expert id
// ---------------------------------------------------------------------------
__global__ void hist_kernel(const int* __restrict__ topE,
                            int* __restrict__ blkCnt, int NKtot) {
  __shared__ int cnt[NE];
  int tid = threadIdx.x;
  if (tid < NE) cnt[tid] = 0;
  __syncthreads();
  int i = blockIdx.x * 256 + tid;
  if (i < NKtot) atomicAdd(&cnt[topE[i]], 1);
  __syncthreads();
  if (tid < NE) blkCnt[blockIdx.x * NE + tid] = cnt[tid];
}

__global__ void scan_kernel(int* __restrict__ blkCnt,
                            int* __restrict__ expertBase, int numBlocks) {
  __shared__ int tot[NE];
  int e = threadIdx.x;
  if (e < NE) {
    int run = 0;
    for (int b = 0; b < numBlocks; ++b) {
      int idx = b * NE + e;
      int t = blkCnt[idx];
      blkCnt[idx] = run;
      run += t;
    }
    tot[e] = run;
  }
  __syncthreads();
  if (e == 0) {
    int acc = 0;
    for (int i = 0; i < NE; ++i) { expertBase[i] = acc; acc += tot[i]; }
    expertBase[NE] = acc;
  }
}

__global__ void scatter_kernel(const int* __restrict__ topE,
                               const float* __restrict__ topG,
                               const int* __restrict__ blkCnt,
                               const int* __restrict__ expertBase,
                               int* __restrict__ srcTok,
                               float* __restrict__ gateS, int NKtot) {
  __shared__ int eid[256];
  int tid = threadIdx.x;
  int b   = blockIdx.x;
  for (int j = tid; j < 256; j += 32) {
    int i = b * 256 + j;
    eid[j] = (i < NKtot) ? topE[i] : -1;
  }
  __syncthreads();
  if (tid < NE) {
    int off = expertBase[tid] + blkCnt[b * NE + tid];
    for (int j = 0; j < 256; ++j) {
      if (eid[j] == tid) {              // preserves original order -> stable
        int i = b * 256 + j;
        srcTok[off] = i >> 2;           // i / K -> token
        gateS[off]  = topG[i];
        ++off;
      }
    }
  }
}

// ---------------------------------------------------------------------------
// Kernel 7: per-expert grouped MLP. block = 128 threads (4 waves), 16 sorted
// rows per tile; wave w owns N-tile w of HD=64. GEMM1 K=1024, GELU, GEMM2 K=64.
// ---------------------------------------------------------------------------
__global__ void expert_mlp_kernel(const float* __restrict__ x,
                                  const __bf16* __restrict__ winPk,
                                  const __bf16* __restrict__ woutPk,
                                  const float* __restrict__ bIn,
                                  const float* __restrict__ bOut,
                                  const int* __restrict__ expertBase,
                                  const int* __restrict__ srcTok,
                                  const float* __restrict__ gateS,
                                  float* __restrict__ out) {
  __shared__ __attribute__((aligned(16))) __bf16 Abf[16 * IN_F]; // 32 KB
  __shared__ __attribute__((aligned(16))) __bf16 Hbf[16 * HD];   // 2 KB
  __shared__ int   stok[16];
  __shared__ float sg[16];
  __shared__ int   sinfo[3];

  const int tid = threadIdx.x;
  if (tid == 0) {
    int bid = blockIdx.x, acc = 0, ex = -1, ti = 0;
    for (int e = 0; e < NE; ++e) {
      int cnt = expertBase[e + 1] - expertBase[e];
      int nt  = (cnt + 15) >> 4;
      if (bid < acc + nt) { ex = e; ti = bid - acc; break; }
      acc += nt;
    }
    sinfo[0] = ex;
    sinfo[1] = (ex >= 0) ? expertBase[ex] + ti * 16 : 0;
    sinfo[2] = (ex >= 0) ? expertBase[ex + 1] : 0;
  }
  __syncthreads();
  const int ex = sinfo[0];
  if (ex < 0) return;                 // grid is an upper bound on tile count
  const int rowBase = sinfo[1], rowEnd = sinfo[2];

  if (tid < 16) {
    int p = rowBase + tid;
    bool valid = p < rowEnd;
    stok[tid] = valid ? srcTok[p] : 0;
    sg[tid]   = valid ? gateS[p] : 0.f;
  }
  __syncthreads();

  // ---- gather + stage A tile (16 rows x 1024) bf16 into LDS ----
  {
    const int row  = tid >> 3;        // 0..15
    const int part = tid & 7;         // 128 floats each
    const float4* src = (const float4*)(x + (size_t)stok[row] * IN_F + part * 128);
    unsigned int* dst = (unsigned int*)&Abf[row * IN_F + part * 128];
#pragma unroll
    for (int i = 0; i < 32; ++i) {
      float4 v = src[i];
      dst[i * 2 + 0] = pack2(v.x, v.y);
      dst[i * 2 + 1] = pack2(v.z, v.w);
    }
  }
  __syncthreads();

  const int w = tid >> 5, lane = tid & 31, hl = lane >> 4, ln = lane & 15;
  const int n = w * 16 + ln;                         // output column (0..63)

  // ---- GEMM1: H = gather(x) @ Win[ex], K = 1024, packed B ----
  const bf16x16* BW = (const bf16x16*)winPk;
  const size_t gBase = (size_t)ex * (IN_F / 32) * HD * 2 + (size_t)n * 2 + hl;
  f32x8 c = {};
  for (int kk = 0; kk < IN_F; kk += 32) {
    const int kk32 = kk >> 5;
    bf16x16 a = ldsA(Abf, ln, IN_F, kk, hl);
    bf16x16 b = BW[gBase + (size_t)kk32 * HD * 2];
    if (kk + 32 < IN_F)
      __builtin_prefetch(&BW[gBase + (size_t)(kk32 + 1) * HD * 2], 0, 0);
    c = wmma_bf16(a, b, c);
  }

  // ---- bias + exact GELU, spill H (bf16) to LDS ----
  {
    float bi = bIn[ex * HD + n];
#pragma unroll
    for (int r = 0; r < 8; ++r) {
      int m = r + 8 * hl;
      float v = c[r] + bi;
      v = 0.5f * v * (1.0f + erff(v * 0.70710678118654752f));
      Hbf[m * HD + n] = (__bf16)v;
    }
  }
  __syncthreads();

  // ---- GEMM2: out = H @ Wout[ex], K = 64 (two 32-steps), packed B ----
  const bf16x16* BQ = (const bf16x16*)woutPk;
  f32x8 c2 = {};
#pragma unroll
  for (int s = 0; s < 2; ++s) {
    bf16x16 a = ldsA(Hbf, ln, HD, s * 32, hl);
    bf16x16 b = BQ[(((size_t)(ex * 2 + s) * HD) + n) * 2 + hl];
    c2 = wmma_bf16(a, b, c2);
  }

  // ---- bias, gate scale, store at sorted position p (out offset p*HD) ----
  {
    float bo = bOut[ex * HD + n];
#pragma unroll
    for (int r = 0; r < 8; ++r) {
      int m = r + 8 * hl;
      int p = rowBase + m;
      if (p < rowEnd)
        out[(size_t)p * HD + n] = (c2[r] + bo) * sg[m];
    }
  }
}

// ---------------------------------------------------------------------------
extern "C" void kernel_launch(void* const* d_in, const int* in_sizes, int n_in,
                              void* d_out, int out_size, void* d_ws, size_t ws_size,
                              hipStream_t stream) {
  (void)n_in; (void)out_size; (void)ws_size;
  const float* x    = (const float*)d_in[0];
  const float* rw   = (const float*)d_in[1];
  const float* hc   = (const float*)d_in[2];
  const float* temp = (const float*)d_in[3];
  const float* Win  = (const float*)d_in[4];
  const float* Wout = (const float*)d_in[5];
  const float* bIn  = (const float*)d_in[6];
  const float* bOut = (const float*)d_in[7];
  float* out = (float*)d_out;

  const int N  = in_sizes[0] / IN_F;          // 16384
  const int NK = N * TK;                      // 65536
  const int histBlocks = (NK + 255) / 256;    // 256

  char* ws = (char*)d_ws;
  size_t off = 0;
  auto wsAlloc = [&](size_t bytes) -> void* {
    void* p = ws + off;
    off = (off + bytes + 255) & ~(size_t)255;
    return p;
  };
  float*  cnn        = (float*) wsAlloc((size_t)NE * HID * sizeof(float));
  int*    topE       = (int*)   wsAlloc((size_t)NK * sizeof(int));
  float*  topG       = (float*) wsAlloc((size_t)NK * sizeof(float));
  int*    blkCnt     = (int*)   wsAlloc((size_t)histBlocks * NE * sizeof(int));
  int*    expertBase = (int*)   wsAlloc((NE + 1) * sizeof(int));
  int*    srcTok     = (int*)   wsAlloc((size_t)NK * sizeof(int));
  float*  gateS      = (float*) wsAlloc((size_t)NK * sizeof(float));
  __bf16* rwPk       = (__bf16*)wsAlloc((size_t)IN_F * HID * 2);        // 512 KB
  __bf16* winPk      = (__bf16*)wsAlloc((size_t)NE * IN_F * HD * 2);    // 2 MB
  __bf16* woutPk     = (__bf16*)wsAlloc((size_t)NE * HD * HD * 2);      // 128 KB
  __bf16* cnnPk      = (__bf16*)wsAlloc((size_t)HID * NE * 2);          // 8 KB

  const int totRw   = (IN_F / 32) * HID * 2;       // 16384 groups
  const int totWin  = NE * (IN_F / 32) * HD * 2;   // 65536 groups
  const int totWout = NE * (HD / 32) * HD * 2;     // 4096 groups
  const int totCnn  = (HID / 32) * NE * 2;         // 256 groups

  prep_centroids_kernel<<<1, 32, 0, stream>>>(hc, temp, cnn);
  pack_b_kernel<<<(totRw + 255) / 256, 256, 0, stream>>>(rw, rwPk, IN_F / 32, HID, totRw);
  pack_b_kernel<<<(totWin + 255) / 256, 256, 0, stream>>>(Win, winPk, IN_F / 32, HD, totWin);
  pack_b_kernel<<<(totWout + 255) / 256, 256, 0, stream>>>(Wout, woutPk, HD / 32, HD, totWout);
  pack_b_kernel<<<(totCnn + 255) / 256, 256, 0, stream>>>(cnn, cnnPk, HID / 32, NE, totCnn);

  router_topk_kernel<<<N / 16, 256, 0, stream>>>(x, rwPk, cnnPk, topE, topG);
  hist_kernel<<<histBlocks, 256, 0, stream>>>(topE, blkCnt, NK);
  scan_kernel<<<1, 32, 0, stream>>>(blkCnt, expertBase, histBlocks);
  scatter_kernel<<<histBlocks, 32, 0, stream>>>(topE, topG, blkCnt, expertBase,
                                                srcTok, gateS, NK);
  expert_mlp_kernel<<<NK / 16 + NE, 128, 0, stream>>>(x, winPk, woutPk, bIn, bOut,
                                                      expertBase, srcTok, gateS, out);
}
